// MuliHeadedMaskedSelfAttention_55697135895166
// MI455X (gfx1250) — matrix operand order repaired and checked by reading, not compiled, not run on hardware
//
#include <hip/hip_runtime.h>
#include <hip/hip_bf16.h>

typedef __attribute__((ext_vector_type(16))) __bf16 v16bf;
typedef __attribute__((ext_vector_type(8)))  __bf16 bf16x8;
typedef __attribute__((ext_vector_type(8)))  float  v8f;

#define Bsz  2
#define Tseq 2048
#define Cdim 768
#define NH   12
#define DK   64

static __device__ __forceinline__ v8f wmma_bf16(v16bf a, v16bf b, v8f c) {
    // D = A(16x32 bf16) * B(32x16 bf16) + C(16x16 f32)
    return __builtin_amdgcn_wmma_f32_16x16x32_bf16(
        /*neg_a=*/false, a, /*neg_b=*/false, b,
        /*c_mod=*/(short)0, c, /*reuse_a=*/false, /*reuse_b=*/false);
}

static __device__ __forceinline__ v16bf cat8(bf16x8 lo, bf16x8 hi) {
    return __builtin_shufflevector(lo, hi, 0, 1, 2, 3, 4, 5, 6, 7,
                                           8, 9, 10, 11, 12, 13, 14, 15);
}

// CDNA5 async copy: 16B global -> LDS, tracked by ASYNCcnt (ISA 15.18.3 op 98).
static __device__ __forceinline__ void async_b128(unsigned lds_byte, const void* g) {
    asm volatile("global_load_async_to_lds_b128 %0, %1, off"
                 :: "v"(lds_byte), "v"((unsigned long long)(uintptr_t)g)
                 : "memory");
}
static __device__ __forceinline__ void wait_async0() {
    asm volatile("s_wait_asynccnt 0" ::: "memory");
}
static __device__ __forceinline__ unsigned lds_off(const void* p) {
    // flat LDS aperture: addr[31:0] is the LDS byte address
    return (unsigned)(uintptr_t)p;
}

// ---------------------------------------------------------------------------
// x (f32) -> bf16, 8 elements/thread, packed 16B stores.
// ---------------------------------------------------------------------------
__global__ __launch_bounds__(256)
void cvt_f32_bf16(const float* __restrict__ in, __bf16* __restrict__ out, int n)
{
    int i = (blockIdx.x * 256 + threadIdx.x) * 8;
    if (i + 8 <= n) {
        bf16x8 v;
        #pragma unroll
        for (int j = 0; j < 8; ++j) v[j] = (__bf16)in[i + j];
        *(bf16x8*)(out + i) = v;
    }
}

// ---------------------------------------------------------------------------
// GEMM: C[M,N] = A[M,K](bf16) * B[K,N](f32 -> bf16), 64x128 block tile,
// 256 thr (8 waves), K in 32-steps. A tile staged via async-to-LDS (raw bf16
// bytes); B tile staged TRANSPOSED (Bs[col][k]) so B fragments are contiguous
// ds_load_b128 pairs. Epilogue modes:
//   0: bf16 scatter  n->(h,d), m->(b,t) into [b,h,t,d]      (Q, K)
//   2: bf16 scatter into [b,h,d,t] (transposed V) -> packed 16B stores
//   1: f32 row-major [M,N]                                   (final output)
// For mode==0 launches, blockIdx.z==2 is promoted to mode 2 (the V matrix).
// ---------------------------------------------------------------------------
__global__ __launch_bounds__(256)
void gemm_bf16_wmma(const __bf16* __restrict__ A,
                    const float* __restrict__ B0, const float* __restrict__ B1,
                    const float* __restrict__ B2,
                    void* O0, void* O1, void* O2,
                    int M, int N, int K, int mode)
{
    const float* Bm = (blockIdx.z == 0) ? B0 : (blockIdx.z == 1 ? B1 : B2);
    void*        Om = (blockIdx.z == 0) ? O0 : (blockIdx.z == 1 ? O1 : O2);
    const int omode = (mode == 1) ? 1 : (blockIdx.z == 2 ? 2 : 0);

    __shared__ __bf16 As[64][40];    // [row][k]   stride 80B (16B aligned)
    __shared__ __bf16 Bs[128][40];   // [col][k]   fragment-major

    const int tid  = threadIdx.x;
    const int lane = tid & 31;
    const int wave = tid >> 5;
    const int wm = wave & 1, wn = wave >> 1;
    const int rowBase = blockIdx.y * 64;
    const int colBase = blockIdx.x * 128;
    const int lr = lane & 15, hi = lane >> 4;

    const int ar   = tid >> 2,  aseg = tid & 3;         // A staging: 64x(4x8)
    const int bcol = tid & 127, bkg  = (tid >> 7) * 16; // B staging: 128x(2x16)

    v8f acc[2][2];
    #pragma unroll
    for (int i = 0; i < 2; ++i)
        #pragma unroll
        for (int j = 0; j < 2; ++j) acc[i][j] = v8f{};

    for (int k0 = 0; k0 < K; k0 += 32) {
        // ---- A tile: raw bf16 bytes, async global->LDS ----
        async_b128(lds_off(&As[ar][aseg * 8]),
                   A + (size_t)(rowBase + ar) * K + k0 + aseg * 8);
        // ---- B tile: coalesced f32 column loads -> bf16 -> Bs[col][k] ----
        {
            const float* src = Bm + (size_t)(k0 + bkg) * N + colBase + bcol;
            bf16x8 p0, p1;
            #pragma unroll
            for (int j = 0; j < 8; ++j) p0[j] = (__bf16)src[(size_t)j * N];
            #pragma unroll
            for (int j = 0; j < 8; ++j) p1[j] = (__bf16)src[(size_t)(j + 8) * N];
            *(bf16x8*)&Bs[bcol][bkg]     = p0;
            *(bf16x8*)&Bs[bcol][bkg + 8] = p1;
        }
        wait_async0();
        __syncthreads();

        // ---- fragments: pure 16B LDS loads (ISA A 16x32 / B 32x16 layouts) ----
        v16bf af[2], bfm[2];
        #pragma unroll
        for (int i = 0; i < 2; ++i) {
            const int row = wm * 32 + i * 16 + lr;
            af[i] = cat8(*(const bf16x8*)&As[row][hi * 8],
                         *(const bf16x8*)&As[row][hi * 8 + 16]);
        }
        #pragma unroll
        for (int j2 = 0; j2 < 2; ++j2) {
            const int col = wn * 32 + j2 * 16 + lr;
            bfm[j2] = cat8(*(const bf16x8*)&Bs[col][hi * 8],
                           *(const bf16x8*)&Bs[col][hi * 8 + 16]);
        }
        #pragma unroll
        for (int i = 0; i < 2; ++i)
            #pragma unroll
            for (int j2 = 0; j2 < 2; ++j2)
                acc[i][j2] = wmma_bf16(af[i], bfm[j2], acc[i][j2]);
        __syncthreads();
    }

    // ---- epilogue (C layout: N = lr, M = hi*8 + r) ----
    #pragma unroll
    for (int i = 0; i < 2; ++i) {
        #pragma unroll
        for (int j2 = 0; j2 < 2; ++j2) {
            const int nIdx  = colBase + wn * 32 + j2 * 16 + lr;
            const int mBase = rowBase + wm * 32 + i * 16 + hi * 8;
            if (omode == 1) {
                float* Of = (float*)Om;
                #pragma unroll
                for (int r = 0; r < 8; ++r)
                    Of[(size_t)(mBase + r) * N + nIdx] = acc[i][j2][r];
            } else if (omode == 0) {
                __bf16* Oh = (__bf16*)Om;
                const int h = nIdx / DK, d = nIdx % DK;
                #pragma unroll
                for (int r = 0; r < 8; ++r) {
                    const int m = mBase + r;
                    const int b = m / Tseq, t = m % Tseq;
                    Oh[(((size_t)(b * NH + h)) * Tseq + t) * DK + d] =
                        (__bf16)acc[i][j2][r];
                }
            } else { // omode == 2: V^T layout [b,h,d,t], t contiguous -> 16B store
                __bf16* Oh = (__bf16*)Om;
                const int h = nIdx / DK, d = nIdx % DK;
                const int b = mBase / Tseq, t0 = mBase % Tseq;
                bf16x8 v;
                #pragma unroll
                for (int r = 0; r < 8; ++r) v[r] = (__bf16)acc[i][j2][r];
                *(bf16x8*)&Oh[(((size_t)(b * NH + h)) * DK + d) * Tseq + t0] = v;
            }
        }
    }
}

// ---------------------------------------------------------------------------
// Flash-style causal attention, one (b,h) per blockIdx.y, 64 queries per
// block (4 waves x 16). S^T = K*Q^T so softmax-over-keys is per-lane (+ one
// shfl_xor(16) across the lane-pair halves of the C layout). V is stored
// transposed [b,h,d,t] (bf16) so the V^T tile is async-copied raw into LDS
// and every fragment is a contiguous 16B load. O written bf16 [b,t,h*64+d].
// ---------------------------------------------------------------------------
__global__ __launch_bounds__(128)
void attn_flash_wmma(const __bf16* __restrict__ Qg, const __bf16* __restrict__ Kg,
                     const __bf16* __restrict__ Vt, __bf16* __restrict__ Og)
{
    __shared__ __bf16 Vts[64][40];   // V^T tile: 64 d-rows x 32 keys, stride 80B

    const int tid  = threadIdx.x;
    const int lane = tid & 31;
    const int wave = tid >> 5;
    const int bh = blockIdx.y;
    const int b = bh / NH, h = bh % NH;
    const int lr = lane & 15, hi = lane >> 4;
    const int qb = blockIdx.x * 64 + wave * 16;
    const int q  = qb + lr;                     // this lane's query (N dim)

    // Q^T B-fragments: rows = d (2 chunks of 32), cols = query; contiguous 16B
    v16bf qf[2];
    {
        const __bf16* qrow = Qg + ((size_t)bh * Tseq + q) * DK;
        #pragma unroll
        for (int c = 0; c < 2; ++c)
            qf[c] = cat8(*(const bf16x8*)(qrow + c * 32 + hi * 8),
                         *(const bf16x8*)(qrow + c * 32 + hi * 8 + 16));
    }

    v8f oacc[4];
    #pragma unroll
    for (int dt = 0; dt < 4; ++dt) oacc[dt] = v8f{};
    float m_run = -1e30f, l_run = 0.0f;

    const int vr = tid >> 1, vhalf = tid & 1;   // V^T staging: 64 rows x 2x32B

    const int nkt = blockIdx.x * 2 + 2;         // causal key range
    for (int kt = 0; kt < nkt; ++kt) {
        const int kb = kt * 32;

        // ---- async stage V^T tile (raw bf16): 64 x 32 ----
        {
            const __bf16* src = Vt + ((size_t)bh * DK + vr) * Tseq + kb + vhalf * 16;
            const unsigned la = lds_off(&Vts[vr][vhalf * 16]);
            async_b128(la, src);
            async_b128(la + 16, src + 8);
        }

        // ---- S^T = K_tile * Q^T (overlaps with async V copy) ----
        float ps[2][8];
        float tmax = -1e30f;
        #pragma unroll
        for (int s = 0; s < 2; ++s) {
            v8f sacc = v8f{};
            #pragma unroll
            for (int c = 0; c < 2; ++c) {
                const __bf16* krow =
                    Kg + ((size_t)bh * Tseq + kb + s * 16 + lr) * DK + c * 32 + hi * 8;
                v16bf kf = cat8(*(const bf16x8*)krow, *(const bf16x8*)(krow + 16));
                sacc = wmma_bf16(kf, qf[c], sacc);
            }
            #pragma unroll
            for (int r = 0; r < 8; ++r) {
                const int key = kb + s * 16 + hi * 8 + r;     // M dim = key
                float sv = sacc[r] * 8.0f;                    // * sqrt(d_k)
                if (key > q) sv = -1e30f;                     // causal mask
                ps[s][r] = sv;
                tmax = fmaxf(tmax, sv);
            }
        }
        tmax = fmaxf(tmax, __shfl_xor(tmax, 16, 32));
        const float m_new = fmaxf(m_run, tmax);
        const float scale = __expf(m_run - m_new);
        float p0[8], p1[8], psum = 0.0f;
        #pragma unroll
        for (int r = 0; r < 8; ++r) {
            p0[r] = __expf(ps[0][r] - m_new);
            p1[r] = __expf(ps[1][r] - m_new);
            psum += p0[r] + p1[r];
        }
        m_run = m_new;
        l_run = l_run * scale + psum;
        #pragma unroll
        for (int dt = 0; dt < 4; ++dt)
            #pragma unroll
            for (int r = 0; r < 8; ++r) oacc[dt][r] *= scale;

        // ---- P^T B-fragment (32 keys x 16 queries) via lane-pair shuffles ----
        v16bf pf;
        #pragma unroll
        for (int j = 0; j < 8; ++j) {
            const float o0 = __shfl_xor(p0[j], 16, 32);
            const float o1 = __shfl_xor(p1[j], 16, 32);
            pf[j]     = (__bf16)((hi == 0) ? p0[j] : o1);
            pf[j + 8] = (__bf16)((hi == 0) ? o0 : p1[j]);
        }

        wait_async0();
        __syncthreads();

        // ---- O^T += V^T * P^T : fragments are contiguous ds_load_b128 ----
        #pragma unroll
        for (int dt = 0; dt < 4; ++dt) {
            const int row = dt * 16 + lr;
            v16bf vf = cat8(*(const bf16x8*)&Vts[row][hi * 16],
                            *(const bf16x8*)&Vts[row][hi * 16 + 8]);
            oacc[dt] = wmma_bf16(vf, pf, oacc[dt]);
        }
        __syncthreads();
    }

    const float l_tot = l_run + __shfl_xor(l_run, 16, 32);
    const float inv = 1.0f / l_tot;
    #pragma unroll
    for (int dt = 0; dt < 4; ++dt) {
        bf16x8 v;
        #pragma unroll
        for (int r = 0; r < 8; ++r) v[r] = (__bf16)(oacc[dt][r] * inv);
        *(bf16x8*)&Og[((size_t)b * Tseq + q) * Cdim + h * DK + dt * 16 + hi * 8] = v;
    }
}

// ---------------------------------------------------------------------------
extern "C" void kernel_launch(void* const* d_in, const int* in_sizes, int n_in,
                              void* d_out, int out_size, void* d_ws, size_t ws_size,
                              hipStream_t stream)
{
    const float* x  = (const float*)d_in[0];
    const float* Wq = (const float*)d_in[1];   // (C, H, D_K) row-major
    const float* Wk = (const float*)d_in[2];
    const float* Wv = (const float*)d_in[3];
    const float* Wo = (const float*)d_in[4];   // (C, C)
    float* out = (float*)d_out;

    const int    M   = Bsz * Tseq;                 // 4096
    const size_t SZ  = (size_t)M * Cdim;           // 3,145,728 elems
    __bf16* wsb = (__bf16*)d_ws;
    __bf16* xb = wsb;              // x in bf16
    __bf16* Qg = wsb + SZ;         // [b,h,t,d]
    __bf16* Kg = wsb + 2 * SZ;     // [b,h,t,d]
    __bf16* Vt = wsb + 3 * SZ;     // [b,h,d,t]  (transposed)
    __bf16* Og = wsb + 4 * SZ;     // [b,t,C]

    // 0) x -> bf16
    cvt_f32_bf16<<<dim3((unsigned)(SZ / (256 * 8))), 256, 0, stream>>>(x, xb, (int)SZ);

    // 1) fused Q/K/V projections (z=2 writes V transposed)
    gemm_bf16_wmma<<<dim3(Cdim / 128, M / 64, 3), 256, 0, stream>>>(
        xb, Wq, Wk, Wv, Qg, Kg, Vt, M, Cdim, Cdim, /*mode=*/0);

    // 2) causal flash attention
    attn_flash_wmma<<<dim3(Tseq / 64, Bsz * NH), 128, 0, stream>>>(Qg, Kg, Vt, Og);

    // 3) output projection (f32 out)
    gemm_bf16_wmma<<<dim3(Cdim / 128, M / 64, 1), 256, 0, stream>>>(
        Og, Wo, Wo, Wo, out, out, out, M, Cdim, Cdim, /*mode=*/1);
}
